// AdaAttN_50766513438721
// MI455X (gfx1250) — matrix operand order repaired and checked by reading, not compiled, not run on hardware
//
#include <hip/hip_runtime.h>

// ---------------------------------------------------------------------------
// AdaAttN on MI455X (gfx1250): bf16 WMMA flash-attention with fused moment
// accumulation, LDS-staged K/V via async global->LDS DMA (double buffered).
//   Q  : [B][N][256] bf16   (query rows, channel-contiguous)  -> B operand
//   Kt : [B][M][256] bf16   (key rows,   channel-contiguous)  -> A operand
//   V  : [B][256][M] bf16   (channel-major)                   -> B operand
//   V2 : [B][256][M] bf16   (V*V)                             -> B operand
// S^T is computed (A=K-tile, B=Q) so lane == query: online softmax is a pure
// per-lane reduction and P lands directly in A-operand layout for P@V, P@V2.
// 4 waves (4 query tiles) per workgroup share each staged K/V chunk.
// ---------------------------------------------------------------------------

typedef __attribute__((ext_vector_type(16))) __bf16 v16bf;
typedef __attribute__((ext_vector_type(8)))  float  v8f;

#define PQ  6400   // spatial positions (80*80)
#define CCH 256    // channels
#define NB  2      // batch
#define NCHUNK (PQ / 32)   // 200 key chunks of 32

// ---- LDS staging layout (bytes), dynamic shared -----------------------------
// K rows padded 256 -> 264 halfs (528 B)  : lane i row stride -> bank 4*i, no conflicts
// V rows padded 32  -> 40  halfs (80 B)   : lane i row stride -> bank 20*i, no conflicts
#define KROWH 264
#define KBUFB (32 * KROWH * 2)          // 16896 B per K buffer
#define VROWH 40
#define VBUFB (256 * VROWH * 2)         // 20480 B per V buffer
#define OFF_K   0
#define OFF_V   (OFF_K + 2 * KBUFB)     // 33792
#define OFF_V2  (OFF_V + 2 * VBUFB)     // 74752
#define SMEM_TOTAL (OFF_V2 + 2 * VBUFB) // 115712 B (of 320 KB/WGP)

// round-to-nearest-even float -> bf16 bits
static __device__ __forceinline__ unsigned short f2bf(float f) {
  union { float f; unsigned u; } c; c.f = f;
  unsigned r = c.u + 0x7FFFu + ((c.u >> 16) & 1u);
  return (unsigned short)(r >> 16);
}

// A-operand (16x32 bf16): per lane, elements 0..7 = halfs [k0..k0+7],
// elements 8..15 = halfs [k0+16..k0+23] (ISA 16-bit A layout).
static __device__ __forceinline__ v16bf load_a16(const unsigned short* row, int k0) {
  v16bf r;
  ((uint4*)&r)[0] = *(const uint4*)(row + k0);
  ((uint4*)&r)[1] = *(const uint4*)(row + k0 + 16);
  return r;
}
// B-operand (32x16 bf16): per lane 16 contiguous halfs (K-contiguous).
static __device__ __forceinline__ v16bf load_b16(const unsigned short* p) {
  v16bf r;
  ((uint4*)&r)[0] = ((const uint4*)p)[0];
  ((uint4*)&r)[1] = ((const uint4*)p)[1];
  return r;
}

// ---- gfx1250 async global->LDS DMA (ASYNCcnt-tracked, no VGPR staging) -----
static __device__ __forceinline__ void async_b128(unsigned lds_addr,
                                                  const unsigned short* g) {
  asm volatile("global_load_async_to_lds_b128 %0, %1, off"
               :: "v"(lds_addr), "v"(g) : "memory");
}
static __device__ __forceinline__ void wait_async0() {
  asm volatile("s_wait_asynccnt 0x0" ::: "memory");
}

// stage one 32-key x 256-ch K chunk (16 KB): thread t -> row t>>2, cols (t&3)+4i
static __device__ __forceinline__ void stage_k(unsigned base,
                                               const unsigned short* kmat,
                                               int kbase, int tid) {
  const int r  = tid >> 2;
  const int c0 = tid & 3;
  const unsigned short* g = kmat + (size_t)(kbase + r) * CCH + c0 * 8;
  const unsigned l = base + (unsigned)r * (KROWH * 2) + (unsigned)c0 * 16;
#pragma unroll
  for (int i = 0; i < 8; ++i)
    async_b128(l + (unsigned)i * 64, g + i * 32);
}
// stage one 256-ch x 32-pos V (or V2) chunk (16 KB): thread t -> rows 2t,2t+1
static __device__ __forceinline__ void stage_v(unsigned base,
                                               const unsigned short* vmat,
                                               int kbase, int tid) {
#pragma unroll
  for (int rr = 0; rr < 2; ++rr) {
    const int row = tid * 2 + rr;
    const unsigned short* g = vmat + (size_t)row * PQ + kbase;
    const unsigned l = base + (unsigned)row * (VROWH * 2);
#pragma unroll
    for (int c = 0; c < 4; ++c)
      async_b128(l + (unsigned)c * 16, g + c * 8);
  }
}

// ---------------------------------------------------------------------------
// Kernel 1: 1x1 conv (GEMM over channels) + bf16 pack into WMMA-friendly
// layouts. vmode 0: out0[b][p][co]; vmode 1: out0[b][co][p], out1 = square.
// ---------------------------------------------------------------------------
__global__ __launch_bounds__(256) void conv1x1_pack(
    const float* __restrict__ x, const float* __restrict__ w,
    const float* __restrict__ bias,
    unsigned short* __restrict__ out0, unsigned short* __restrict__ out1,
    int vmode) {
  __shared__ float lds[CCH][8];          // 8 positions x 256 channels (8 KB)
  const int t     = threadIdx.x;         // out channel / staging channel
  const int b     = blockIdx.y;
  const int pbase = blockIdx.x * 8;

  const float* src = x + ((size_t)b * CCH + t) * PQ + pbase;
  float4 v0 = ((const float4*)src)[0];
  float4 v1 = ((const float4*)src)[1];
  lds[t][0] = v0.x; lds[t][1] = v0.y; lds[t][2] = v0.z; lds[t][3] = v0.w;
  lds[t][4] = v1.x; lds[t][5] = v1.y; lds[t][6] = v1.z; lds[t][7] = v1.w;
  __syncthreads();

  float acc[8];
  const float bb = bias[t];
#pragma unroll
  for (int p = 0; p < 8; ++p) acc[p] = bb;

  const float* wr = w + (size_t)t * CCH;
  for (int ci = 0; ci < CCH; ++ci) {
    const float wv = wr[ci];
#pragma unroll
    for (int p = 0; p < 8; ++p) acc[p] = fmaf(wv, lds[ci][p], acc[p]);
  }

  if (vmode == 0) {
#pragma unroll
    for (int p = 0; p < 8; ++p)
      out0[((size_t)b * PQ + pbase + p) * CCH + t] = f2bf(acc[p]);
  } else {
    const size_t base = ((size_t)b * CCH + t) * PQ + pbase;
#pragma unroll
    for (int p = 0; p < 8; ++p) {
      float v = acc[p];
      out0[base + p] = f2bf(v);
      out1[base + p] = f2bf(v * v);
    }
  }
}

// ---------------------------------------------------------------------------
// Kernel 2: flash attention. 128 threads = 4 waves, one 16-query tile each;
// the workgroup cooperatively double-buffers K/V/V2 chunks in LDS via
// global_load_async_to_lds_b128.
// ---------------------------------------------------------------------------
__global__ __launch_bounds__(128) void adaattn_flash(
    const unsigned short* __restrict__ Qb, const unsigned short* __restrict__ Kt,
    const unsigned short* __restrict__ Vb, const unsigned short* __restrict__ V2,
    float* __restrict__ meanOut, float* __restrict__ secOut) {
  extern __shared__ unsigned char smem[];
  const int tid   = threadIdx.x;
  const int lane  = tid & 31;
  const int wave  = tid >> 5;
  const int b     = blockIdx.y;
  const int qbase = (blockIdx.x * 4 + wave) * 16;
  const int l15   = lane & 15;
  const int hi    = (lane >= 16) ? 1 : 0;

  // low 32 bits of the generic LDS pointer == LDS byte address
  const unsigned smem_base = (unsigned)(size_t)(void*)smem;

  const unsigned short* kmat  = Kt + (size_t)b * PQ * CCH;
  const unsigned short* vmat  = Vb + (size_t)b * CCH * PQ;
  const unsigned short* v2mat = V2 + (size_t)b * CCH * PQ;

  // resident Q fragments (B operand), one per 32-channel chunk
  v16bf qreg[8];
  const unsigned short* qrow = Qb + ((size_t)b * PQ + qbase + l15) * CCH;
#pragma unroll
  for (int kk = 0; kk < 8; ++kk)
    qreg[kk] = load_a16(qrow, kk * 32 + (hi ? 8 : 0));

  // ---------------- pass 1: online max / Z ----------------
  stage_k(smem_base + OFF_K, kmat, 0, tid);

  float m = -1e30f, Z = 0.f;
  for (int kc = 0; kc < NCHUNK; ++kc) {
    const int cur = kc & 1;
    wait_async0();
    __syncthreads();                         // chunk kc resident for all waves
    if (kc + 1 < NCHUNK)
      stage_k(smem_base + OFF_K + (cur ^ 1) * KBUFB, kmat, (kc + 1) * 32, tid);

    const unsigned short* kbuf = (const unsigned short*)(smem + OFF_K + cur * KBUFB);
    const unsigned short* kr0  = kbuf + (size_t)l15 * KROWH;
    const unsigned short* kr1  = kr0 + (size_t)16 * KROWH;
    v8f s0 = {}, s1 = {};
#pragma unroll
    for (int kk = 0; kk < 8; ++kk) {
      const int k0 = kk * 32 + (hi ? 8 : 0);
      s0 = __builtin_amdgcn_wmma_f32_16x16x32_bf16(
          false, load_a16(kr0, k0), false, qreg[kk], (short)0, s0, false, false);
      s1 = __builtin_amdgcn_wmma_f32_16x16x32_bf16(
          false, load_a16(kr1, k0), false, qreg[kk], (short)0, s1, false, false);
    }
    float mt = s0[0];
#pragma unroll
    for (int j = 0; j < 8; ++j) { mt = fmaxf(mt, s0[j]); mt = fmaxf(mt, s1[j]); }
    const float mn = fmaxf(m, mt);
    float zadd = 0.f;
#pragma unroll
    for (int j = 0; j < 8; ++j)
      zadd += __expf(s0[j] - mn) + __expf(s1[j] - mn);
    Z = Z * __expf(m - mn) + zadd;
    m = mn;
  }
  // each half-wave saw half of every key chunk: merge (lane q <-> q+16)
  {
    const float mo = __shfl_xor(m, 16, 32);
    const float Zo = __shfl_xor(Z, 16, 32);
    const float mc = fmaxf(m, mo);
    Z = Z * __expf(m - mc) + Zo * __expf(mo - mc);
    m = mc;
  }

  // ---------------- pass 2: accumulate moments ----------------
  v8f macc[16], sacc[16];
#pragma unroll
  for (int cs = 0; cs < 16; ++cs) { macc[cs] = (v8f){}; sacc[cs] = (v8f){}; }

  __syncthreads();                           // pass-1 reads of buf0 finished
  stage_k(smem_base + OFF_K,  kmat,  0, tid);
  stage_v(smem_base + OFF_V,  vmat,  0, tid);
  stage_v(smem_base + OFF_V2, v2mat, 0, tid);

  for (int kc = 0; kc < NCHUNK; ++kc) {
    const int cur = kc & 1;
    wait_async0();
    __syncthreads();
    if (kc + 1 < NCHUNK) {
      const int nb = (cur ^ 1);
      const int nk = (kc + 1) * 32;
      stage_k(smem_base + OFF_K  + nb * KBUFB, kmat,  nk, tid);
      stage_v(smem_base + OFF_V  + nb * VBUFB, vmat,  nk, tid);
      stage_v(smem_base + OFF_V2 + nb * VBUFB, v2mat, nk, tid);
    }

    const unsigned short* kbuf = (const unsigned short*)(smem + OFF_K + cur * KBUFB);
    const unsigned short* kr0  = kbuf + (size_t)l15 * KROWH;
    const unsigned short* kr1  = kr0 + (size_t)16 * KROWH;
    v8f s0 = {}, s1 = {};
#pragma unroll
    for (int kk = 0; kk < 8; ++kk) {
      const int k0 = kk * 32 + (hi ? 8 : 0);
      s0 = __builtin_amdgcn_wmma_f32_16x16x32_bf16(
          false, load_a16(kr0, k0), false, qreg[kk], (short)0, s0, false, false);
      s1 = __builtin_amdgcn_wmma_f32_16x16x32_bf16(
          false, load_a16(kr1, k0), false, qreg[kk], (short)0, s1, false, false);
    }
    // P fragment: directly in 16-bit A-operand layout (lane = query row)
    v16bf pa;
#pragma unroll
    for (int j = 0; j < 8; ++j) {
      pa[j]     = (__bf16)__expf(s0[j] - m);
      pa[8 + j] = (__bf16)__expf(s1[j] - m);
    }
    const unsigned short* vrow  =
        (const unsigned short*)(smem + OFF_V  + cur * VBUFB) + (size_t)l15 * VROWH + (hi ? 8 : 0);
    const unsigned short* vrow2 =
        (const unsigned short*)(smem + OFF_V2 + cur * VBUFB) + (size_t)l15 * VROWH + (hi ? 8 : 0);
#pragma unroll
    for (int cs = 0; cs < 16; ++cs) {
      const v16bf vb  = load_a16(vrow  + (size_t)cs * 16 * VROWH, 0);
      const v16bf vb2 = load_a16(vrow2 + (size_t)cs * 16 * VROWH, 0);
      macc[cs] = __builtin_amdgcn_wmma_f32_16x16x32_bf16(
          false, pa, false, vb,  (short)0, macc[cs], false, false);
      sacc[cs] = __builtin_amdgcn_wmma_f32_16x16x32_bf16(
          false, pa, false, vb2, (short)0, sacc[cs], false, false);
    }
  }

  // normalize by Z and store. D layout: lane -> channel (l&15),
  // vgpr j -> query j + 8*hi within the tile.
  float rz[8];
#pragma unroll
  for (int j = 0; j < 8; ++j)
    rz[j] = 1.f / __shfl(Z, j + 8 * hi, 32);

#pragma unroll
  for (int cs = 0; cs < 16; ++cs) {
    const int ch = cs * 16 + l15;
#pragma unroll
    for (int j = 0; j < 8; ++j) {
      const int q = qbase + j + 8 * hi;
      const size_t o = ((size_t)b * CCH + ch) * PQ + q;
      meanOut[o] = macc[cs][j] * rz[j];
      secOut[o]  = sacc[cs][j] * rz[j];
    }
  }
}

// ---------------------------------------------------------------------------
// Kernel 3: instance-norm of content + combine:
//   out = sqrt(relu(second - mean^2)) * (content - mu)*rsqrt(var+eps) + mean
// ---------------------------------------------------------------------------
__global__ __launch_bounds__(256) void finalize_norm(
    const float* __restrict__ content, const float* __restrict__ meanIn,
    const float* __restrict__ secIn, float* __restrict__ out) {
  __shared__ float red[256];
  __shared__ float red2[256];
  const int t = threadIdx.x;
  const int c = blockIdx.x;
  const int b = blockIdx.y;
  const size_t base = ((size_t)b * CCH + c) * PQ;

  float s = 0.f, s2 = 0.f;
  for (int p = t; p < PQ; p += 256) {
    const float v = content[base + p];
    s += v; s2 += v * v;
  }
  red[t] = s; red2[t] = s2;
  __syncthreads();
  for (int off = 128; off > 0; off >>= 1) {
    if (t < off) { red[t] += red[t + off]; red2[t] += red2[t + off]; }
    __syncthreads();
  }
  const float mu   = red[0] * (1.f / PQ);
  const float var  = red2[0] * (1.f / PQ) - mu * mu;
  const float rstd = rsqrtf(var + 1e-5f);

  for (int p = t; p < PQ; p += 256) {
    const float mval = meanIn[base + p];
    const float sval = secIn[base + p];
    const float sd   = sqrtf(fmaxf(sval - mval * mval, 0.f));
    out[base + p] = sd * (content[base + p] - mu) * rstd + mval;
  }
}

// ---------------------------------------------------------------------------
extern "C" void kernel_launch(void* const* d_in, const int* in_sizes, int n_in,
                              void* d_out, int out_size, void* d_ws, size_t ws_size,
                              hipStream_t stream) {
  const float* content     = (const float*)d_in[0];
  const float* style       = (const float*)d_in[1];
  const float* content_key = (const float*)d_in[2];
  const float* style_key   = (const float*)d_in[3];
  const float* f_w = (const float*)d_in[4];
  const float* f_b = (const float*)d_in[5];
  const float* g_w = (const float*)d_in[6];
  const float* g_b = (const float*)d_in[7];
  const float* h_w = (const float*)d_in[8];
  const float* h_b = (const float*)d_in[9];

  // workspace layout (~52.4 MB total)
  const size_t PC = (size_t)NB * PQ * CCH;                  // 3,276,800 elems
  unsigned short* Qb = (unsigned short*)d_ws;               // bf16 [B][N][C]
  unsigned short* Kt = Qb + PC;                             // bf16 [B][M][C]
  unsigned short* Vb = Kt + PC;                             // bf16 [B][C][M]
  unsigned short* V2 = Vb + PC;                             // bf16 [B][C][M]
  float* meanW = (float*)(V2 + PC);                         // f32  [B][C][N]
  float* secW  = meanW + PC;                                // f32  [B][C][N]

  dim3 cgrid(PQ / 8, NB);
  conv1x1_pack<<<cgrid, 256, 0, stream>>>(content_key, f_w, f_b, Qb, nullptr, 0);
  conv1x1_pack<<<cgrid, 256, 0, stream>>>(style_key,   g_w, g_b, Kt, nullptr, 0);
  conv1x1_pack<<<cgrid, 256, 0, stream>>>(style,       h_w, h_b, Vb, V2,      1);

  dim3 fgrid(PQ / 64, NB);   // 4 query tiles per workgroup
  adaattn_flash<<<fgrid, 128, SMEM_TOTAL, stream>>>(Qb, Kt, Vb, V2, meanW, secW);

  dim3 ngrid(CCH, NB);
  finalize_norm<<<ngrid, 256, 0, stream>>>(content, meanW, secW, (float*)d_out);
}
